// SynthesisLayer3_69037304315915
// MI455X (gfx1250) — matrix operand order repaired and checked by reading, not compiled, not run on hardware
//
#include <hip/hip_runtime.h>

typedef __attribute__((ext_vector_type(16))) __bf16 v16bf;
typedef __attribute__((ext_vector_type(8)))  float  v8f;
typedef __attribute__((__vector_size__(16))) int i32x4;
typedef __attribute__((address_space(1))) i32x4 gi32x4;   // global int4
typedef __attribute__((address_space(3))) i32x4 li32x4;   // LDS int4

#define CINC   512
#define COUTC  512
#define BATCH  8
#define HW     64
#define HCONV  66
#define MCONV  (HCONV*HCONV)      /* 4356 */
#define NPIX   (HW*HW)            /* 4096 */
#define NXTOT  (BATCH*CINC*NPIX)  /* 16777216 */

#if defined(__has_builtin)
# if __has_builtin(__builtin_amdgcn_global_load_async_to_lds_b128)
#  define ASYNC_LDS 1
# endif
#endif
#ifndef ASYNC_LDS
# define ASYNC_LDS 0
#endif

__device__ __forceinline__ __bf16 f2bf(float f) { return (__bf16)f; }

// ---------------- stage 0: zero pad scratch (async-load target for halo lanes) -------------
__global__ void k_zero(float* __restrict__ z) { z[threadIdx.x] = 0.f; }

// ---------------- stage 1: mean(x^2) partial reduction ----------------
__global__ void k_sq_partial(const float* __restrict__ x, float* __restrict__ part) {
  __shared__ float red[256];
  float s = 0.f;
  for (int i = blockIdx.x * 256 + threadIdx.x; i < NXTOT; i += gridDim.x * 256) {
    float v = x[i];
    s += v * v;
  }
  red[threadIdx.x] = s;
  __syncthreads();
  for (int w = 128; w > 0; w >>= 1) {
    if ((int)threadIdx.x < w) red[threadIdx.x] += red[threadIdx.x + w];
    __syncthreads();
  }
  if (threadIdx.x == 0) part[blockIdx.x] = red[0];
}

__global__ void k_finalize(const float* __restrict__ part, const float* __restrict__ w_avg,
                           float* __restrict__ scale) {
  __shared__ float red[256];
  float s = part[threadIdx.x] + part[threadIdx.x + 256] +
            part[threadIdx.x + 512] + part[threadIdx.x + 768];
  red[threadIdx.x] = s;
  __syncthreads();
  for (int w = 128; w > 0; w >>= 1) {
    if ((int)threadIdx.x < w) red[threadIdx.x] += red[threadIdx.x + w];
    __syncthreads();
  }
  if (threadIdx.x == 0) {
    float mag = red[0] * (1.0f / (float)NXTOT);
    float wnew = mag + 0.999f * (w_avg[0] - mag);  // lerp(mag, w_avg, beta)
    scale[0] = rsqrtf(wnew);
  }
}

// ---------------- stage 2: styles = w @ (affine_w/sqrt(512)).T + affine_b ----------------
__global__ void k_styles(const float* __restrict__ w, const float* __restrict__ aw,
                         const float* __restrict__ ab, float* __restrict__ styles) {
  int idx = blockIdx.x * 256 + threadIdx.x;       // 8*512
  int b = idx >> 9, i = idx & 511;
  const float* wr = w + b * 512;
  const float* ar = aw + i * 512;
  float acc = 0.f;
  for (int d = 0; d < 512; ++d) acc += wr[d] * ar[d];
  styles[idx] = acc * 0.044194173824159216f + ab[i];   // 1/sqrt(512)
}

// ---------------- stage 3: wsq[o][i] = sum_k conv_w[o,i,k]^2 ----------------
__global__ void k_wsq(const float* __restrict__ cw, float* __restrict__ wsq) {
  int idx = blockIdx.x * 256 + threadIdx.x;       // 512*512
  int o = idx >> 9, i = idx & 511;
  const float* p = cw + (size_t)(o * 512 + i) * 9;
  float s = 0.f;
  for (int k = 0; k < 9; ++k) s += p[k] * p[k];
  wsq[idx] = s;
}

// ---------------- stage 4: demod[b][o] = rsqrt(sum_i wsq[o,i]*styles[b,i]^2 + eps) ----------
__global__ void k_demod(const float* __restrict__ styles, const float* __restrict__ wsq,
                        float* __restrict__ demod) {
  int idx = blockIdx.x * 256 + threadIdx.x;       // 8*512
  int b = idx >> 9, o = idx & 511;
  const float* sr = styles + b * 512;
  const float* wr = wsq + o * 512;
  float acc = 0.f;
  for (int i = 0; i < 512; ++i) { float s = sr[i]; acc += wr[i] * s * s; }
  demod[idx] = rsqrtf(acc + 1e-8f);
}

// ---------------- stage 5: pack x to modulated NHWC bf16 (LDS tile transpose) -------------
__global__ void __launch_bounds__(256) k_pack_x(const float* __restrict__ x,
                                                const float* __restrict__ styles,
                                                const float* __restrict__ scale,
                                                __bf16* __restrict__ xm) {
  __shared__ float ts[32][65];
  const int b  = blockIdx.z;
  const int i0 = blockIdx.y * 32;
  const int p0 = blockIdx.x * 64;
  const int t  = threadIdx.x;
  #pragma unroll
  for (int e = 0; e < 8; ++e) {
    const int idx = e * 256 + t;
    const int il = idx >> 6, pl = idx & 63;
    ts[il][pl] = x[((size_t)(b * 512 + i0 + il)) * NPIX + p0 + pl];
  }
  __syncthreads();
  const float sc = scale[0];
  #pragma unroll
  for (int e = 0; e < 8; ++e) {
    const int idx = e * 256 + t;
    const int pl = idx >> 5, cl = idx & 31;
    xm[((size_t)(b * NPIX + p0 + pl)) * CINC + i0 + cl] =
        f2bf(ts[cl][pl] * sc * styles[b * 512 + i0 + cl]);
  }
}

// ---------------- stage 6: pack weights into per-lane WMMA B-fragment order ----------------
// wfrag[((rk*16 + icb)*512 + n)*32 + k] = conv_w[n][icb*32+k][rk]
__global__ void k_pack_w(const float* __restrict__ cw, __bf16* __restrict__ wp) {
  int idx = blockIdx.x * 256 + threadIdx.x;       // 9*16*512*32 = 2359296
  int k   = idx & 31;
  int n   = (idx >> 5) & 511;
  int icb = (idx >> 14) & 15;
  int rk  = idx >> 18;
  wp[idx] = f2bf(cw[(size_t)(n * 512 + icb * 32 + k) * 9 + rk]);
}

// ---------------- stage 7: implicit-GEMM 3x3 conv via WMMA bf16 ----------------
// Block tile: 128 pixels x 128 out-channels; wave tile 64x32 (8 WMMAs / K-step).
// A tile double-buffered in LDS, staged with GLOBAL_LOAD_ASYNC_TO_LDS_B128
// (halo lanes redirected to a zeroed scratch line so EXEC stays all-ones); B fragments
// read directly from L2-resident pre-swizzled weights.
#define ASTRIDE 40   /* bf16 elems; 80B rows: 16B-aligned, conflict-free frag reads */
__global__ void __launch_bounds__(256) k_gemm(const __bf16* __restrict__ xmod,
                                              const __bf16* __restrict__ wfrag,
                                              const float* __restrict__ demod,
                                              const float* __restrict__ bias,
                                              const float* __restrict__ zpad,
                                              float* __restrict__ ycnv) {
  __shared__ __bf16 a_lds[2][128][ASTRIDE];

  const int b     = blockIdx.z;
  const int mtile = blockIdx.x;            // 0..34 (128 pixels each)
  const int tid   = threadIdx.x;
  const int wave  = tid >> 5;
  const int lane  = tid & 31;
  const int wm    = wave >> 2;             // 0..1: M sub-tile (64 rows)
  const int wn    = wave & 3;              // 0..3: N sub-tile (32 cols = 2 frag tiles)

  const int ap  = tid >> 1;                // staged pixel
  const int acp = (tid & 1) * 16;          // channel sub-offset (32B half-row)
  const int mg  = mtile * 128 + ap;
  const int oh  = mg / HCONV;
  const int ow  = mg - oh * HCONV;

  const int mrow  = lane & 15;
  const int half  = lane >> 4;
  const int kbase = half ? 8 : 0;          // ISA 16-bit A-frag K base
  const int ncol  = lane & 15;
  const int n0    = blockIdx.y * 128 + wn * 32 + ncol;   // frag0 col; frag1 = n0+16

  v8f acc[2][4];
  #pragma unroll
  for (int nt = 0; nt < 2; ++nt)
    #pragma unroll
    for (int mt = 0; mt < 4; ++mt)
      #pragma unroll
      for (int e = 0; e < 8; ++e) acc[nt][mt][e] = 0.f;

  // 144 K-steps: it = rk*16 + icb  (rk: 3x3 tap, icb: 32-channel block)
  auto aSrc = [&](int it) -> const __bf16* {
    const int rk = it >> 4;
    const int ic = (it & 15) << 5;
    const int kh = rk / 3, kw = rk - 3 * kh;
    const int ih = oh + kh - 2;
    const int iw = ow + kw - 2;
    const bool valid =
        (mg < MCONV) & ((unsigned)ih < (unsigned)HW) & ((unsigned)iw < (unsigned)HW);
    return valid ? xmod + ((size_t)(b * NPIX + ih * HW + iw)) * CINC + ic + acp
                 : (const __bf16*)zpad;
  };

  auto stageA = [&](int it, int buf) {
    const __bf16* src = aSrc(it);
    __bf16* dst = &a_lds[buf][ap][acp];
#if ASYNC_LDS
    // generic LDS address low 32 bits == LDS byte offset (ISA 10.2 aperture rules)
    __builtin_amdgcn_global_load_async_to_lds_b128(
        (gi32x4*)(uintptr_t)src,
        (li32x4*)(uintptr_t)(uint32_t)(uintptr_t)dst, 0, 0);
    __builtin_amdgcn_global_load_async_to_lds_b128(
        (gi32x4*)(uintptr_t)src,
        (li32x4*)(uintptr_t)(uint32_t)(uintptr_t)dst, 16, 0);
#else
    const uint4* s = reinterpret_cast<const uint4*>(src);
    uint4 lo = s[0], hi = s[1];
    *reinterpret_cast<uint4*>(dst)     = lo;
    *reinterpret_cast<uint4*>(dst + 8) = hi;
#endif
  };

  uint4 bq[4];
  auto fetchB = [&](int it) {
    const uint4* s0 = reinterpret_cast<const uint4*>(
        wfrag + (((size_t)it * 512 + n0) * 32));
    const uint4* s1 = reinterpret_cast<const uint4*>(
        wfrag + (((size_t)it * 512 + n0 + 16) * 32));
    bq[0] = s0[half]; bq[1] = s0[2 + half];
    bq[2] = s1[half]; bq[3] = s1[2 + half];
  };

  stageA(0, 0);
  fetchB(0);

  for (int it = 0; it < 144; ++it) {
    const int buf = it & 1;
#if ASYNC_LDS
    asm volatile("s_wait_asynccnt 0x0" ::: "memory");   // my async stores to LDS landed
#endif
    union { v16bf v; uint4 q[2]; } bf0, bf1;
    bf0.q[0] = bq[0]; bf0.q[1] = bq[1];
    bf1.q[0] = bq[2]; bf1.q[1] = bq[3];
    __syncthreads();                                    // tile 'it' published; buf^1 free
    if (it + 1 < 144) {                                 // stage next tile behind compute
      stageA(it + 1, buf ^ 1);
      fetchB(it + 1);
    }

    #pragma unroll
    for (int mt = 0; mt < 4; ++mt) {
      const unsigned int* arow =
          reinterpret_cast<const unsigned int*>(&a_lds[buf][wm * 64 + mt * 16 + mrow][0]);
      union { v16bf v; unsigned int u[8]; } af;
      #pragma unroll
      for (int v = 0; v < 4; ++v) af.u[v] = arow[(kbase >> 1) + v];
      #pragma unroll
      for (int v = 0; v < 4; ++v) af.u[4 + v] = arow[((kbase + 16) >> 1) + v];

      acc[0][mt] = __builtin_amdgcn_wmma_f32_16x16x32_bf16(
          false, af.v, false, bf0.v, (short)0, acc[0][mt], false, false);
      acc[1][mt] = __builtin_amdgcn_wmma_f32_16x16x32_bf16(
          false, af.v, false, bf1.v, (short)0, acc[1][mt], false, false);
    }
  }

  // ---- epilogue: demod + bias, store NHWC f32 ----
  #pragma unroll
  for (int nt = 0; nt < 2; ++nt) {
    const int og = n0 + nt * 16;
    const float dd = demod[b * 512 + og];
    const float bb = bias[og];
    #pragma unroll
    for (int mt = 0; mt < 4; ++mt) {
      #pragma unroll
      for (int r = 0; r < 8; ++r) {
        const int m = mtile * 128 + wm * 64 + mt * 16 + half * 8 + r;
        if (m < MCONV)
          ycnv[((size_t)(b * MCONV + m)) * COUTC + og] = acc[nt][mt][r] * dd + bb;
      }
    }
  }
}

// ---------------- stage 8: fused up-FIR -> lrelu/clamp -> down-FIR ----------------
#define NCF 8
__global__ void __launch_bounds__(256) k_upfirdn(const float* __restrict__ yc,
                                                 const float* __restrict__ fup,
                                                 const float* __restrict__ fdn,
                                                 float* __restrict__ out) {
  __shared__ float smem[9152];
  float* vt = smem;           // [26][18][8]  vertical-up result
  float* at = smem + 3744;    // [26][26][8]  activated up samples
  float* xs = smem + 3744;    // [18][18][8]  input tile (dead before at is written)
  float* dt = smem;           // [26][8][8]   horizontal-down result (vt dead by then)

  const int b   = blockIdx.z;
  const int c0  = blockIdx.y * NCF;
  const int st  = blockIdx.x;             // 0..63
  const int ty0 = (st >> 3) * 8;
  const int tx0 = (st & 7) * 8;
  const int c   = threadIdx.x & (NCF - 1);
  const int q   = threadIdx.x >> 3;       // 0..31 workers per channel

  float FU[12], FD[12];
  #pragma unroll
  for (int t = 0; t < 12; ++t) { FU[t] = fup[t]; FD[t] = fdn[t]; }

  for (int pos = q; pos < 324; pos += 32) {
    const int r = pos / 18, cl = pos - r * 18;
    const int jh = ty0 - 4 + r, jw = tx0 - 4 + cl;
    float v = 0.f;
    if ((unsigned)jh < (unsigned)HCONV && (unsigned)jw < (unsigned)HCONV)
      v = yc[(((size_t)b * MCONV + jh * HCONV + jw) << 9) + c0 + c];
    xs[(r * 18 + cl) * NCF + c] = v;
  }
  __syncthreads();

  for (int pos = q; pos < 468; pos += 32) {
    const int uyl = pos / 18, cl = pos - uyl * 18;
    const int uy = 2 * ty0 + uyl;
    const int t0 = (uy + 9) & 1;
    float a = 0.f;
    #pragma unroll
    for (int tt = 0; tt < 6; ++tt) {
      const int t = t0 + 2 * tt;
      const int jl = ((uy + t - 9) >> 1) - ty0 + 4;
      a += FU[t] * xs[(jl * 18 + cl) * NCF + c];
    }
    vt[(uyl * 18 + cl) * NCF + c] = a;
  }
  __syncthreads();

  for (int pos = q; pos < 676; pos += 32) {
    const int uyl = pos / 26, uxl = pos - uyl * 26;
    const int ux = 2 * tx0 + uxl;
    const int t0 = (ux + 9) & 1;
    float a = 0.f;
    #pragma unroll
    for (int tt = 0; tt < 6; ++tt) {
      const int t = t0 + 2 * tt;
      const int jl = ((ux + t - 9) >> 1) - tx0 + 4;
      a += FU[t] * vt[(uyl * 18 + jl) * NCF + c];
    }
    float v = 4.0f * a;                                  // upfirdn gain = UP^2
    v = (v >= 0.f ? v : 0.2f * v) * 1.4142135623730951f; // lrelu + sqrt(2) gain
    v = fminf(fmaxf(v, -256.f), 256.f);                  // clamp
    at[(uyl * 26 + uxl) * NCF + c] = v;
  }
  __syncthreads();

  for (int pos = q; pos < 208; pos += 32) {
    const int uyl = pos >> 3, oxl = pos & 7;
    float a = 0.f;
    #pragma unroll
    for (int t = 0; t < 12; ++t)
      a += FD[t] * at[(uyl * 26 + 2 * oxl + t) * NCF + c];
    dt[(uyl * 8 + oxl) * NCF + c] = a;
  }
  __syncthreads();

  for (int pos = q; pos < 64; pos += 32) {
    const int oyl = pos >> 3, oxl = pos & 7;
    float a = 0.f;
    #pragma unroll
    for (int t = 0; t < 12; ++t)
      a += FD[t] * dt[((2 * oyl + t) * 8 + oxl) * NCF + c];
    out[(((size_t)b * COUTC + c0 + c) * HW + ty0 + oyl) * HW + tx0 + oxl] = a;
  }
}

// ---------------- host side ----------------
extern "C" void kernel_launch(void* const* d_in, const int* in_sizes, int n_in,
                              void* d_out, int out_size, void* d_ws, size_t ws_size,
                              hipStream_t stream) {
  const float* x    = (const float*)d_in[0];   // [8,512,64,64]
  const float* w    = (const float*)d_in[1];   // [8,512]
  const float* aw   = (const float*)d_in[2];   // [512,512]
  const float* ab   = (const float*)d_in[3];   // [512]
  const float* cw   = (const float*)d_in[4];   // [512,512,3,3]
  const float* cb   = (const float*)d_in[5];   // [512]
  const float* fup  = (const float*)d_in[6];   // [12]
  const float* fdn  = (const float*)d_in[7];   // [12]
  const float* wavg = (const float*)d_in[8];   // [1]
  float* outp = (float*)d_out;

  char* ws = (char*)d_ws;
  float*  part   = (float*)(ws + 0);                      //   4 KB (1024 floats)
  float*  scale  = (float*)(ws + 4096);                   //   scalar
  float*  zpad   = (float*)(ws + 4352);                   //   256 B zeros (halo target)
  float*  styles = (float*)(ws + 4608);                   //  16 KB
  float*  wsq    = (float*)(ws + 20992);                  //   1 MB
  float*  demod  = (float*)(ws + 1069568);                //  16 KB
  __bf16* xmod   = (__bf16*)(ws + 1085952);               //  32 MB
  __bf16* wfrag  = (__bf16*)(ws + 34640384);              // 4.5 MB
  float*  ycnv   = (float*)(ws + 39358976);               //  68 MB  (ends ~105.6 MB)

  k_zero      <<<1,     64, 0, stream>>>(zpad);
  k_sq_partial<<<1024, 256, 0, stream>>>(x, part);
  k_finalize  <<<1,    256, 0, stream>>>(part, wavg, scale);
  k_styles    <<<16,   256, 0, stream>>>(w, aw, ab, styles);
  k_wsq       <<<1024, 256, 0, stream>>>(cw, wsq);
  k_demod     <<<16,   256, 0, stream>>>(styles, wsq, demod);
  k_pack_x    <<<dim3(64, 16, 8), 256, 0, stream>>>(x, styles, scale, xmod);
  k_pack_w    <<<9216, 256, 0, stream>>>(cw, wfrag);
  k_gemm      <<<dim3(35, 4, 8),  256, 0, stream>>>(xmod, wfrag, demod, cb, zpad, ycnv);
  k_upfirdn   <<<dim3(64, 64, 8), 256, 0, stream>>>(ycnv, fup, fdn, outp);
}
